// MoELayer_43928925503871
// MI455X (gfx1250) — compile-verified, matrix operand order checked by reading, not run on hardware
//
#include <hip/hip_runtime.h>

typedef __attribute__((ext_vector_type(16))) __bf16 v16bf;
typedef __attribute__((ext_vector_type(8)))  float  v8f;

constexpr int T   = 4096;   // B*S tokens
constexpr int HD  = 768;    // hidden
constexpr int FI  = 3072;   // inter
constexpr int NE  = 8;      // experts
constexpr int MT  = 64;     // token tile (rows) per workgroup
constexpr int FC  = 32;     // inter chunk width
constexpr int NTHR = 512;   // 16 waves
constexpr int XS  = HD + 4; // LDS strides (bf16 elements)
constexpr int W1S = HD + 4;
constexpr int W2S = FC + 4;
constexpr int HCS = FC + 4;
constexpr int MBLK = T / MT; // 64 row-tiles per expert (capacity)

// ---------------------------------------------------------------- zero output
__global__ void moe_zero(float* __restrict__ out, int n) {
  int i = blockIdx.x * blockDim.x + threadIdx.x;
  if (i < n) out[i] = 0.f;
}

// ---------------------------------------------------------------- routing
__global__ void moe_route(const float* __restrict__ x, const float* __restrict__ Wg,
                          int* __restrict__ ridx, float* __restrict__ rw) {
  __shared__ float wgs[HD * NE];
  for (int i = threadIdx.x; i < HD * NE; i += blockDim.x) wgs[i] = Wg[i];
  __syncthreads();
  int t = blockIdx.x * blockDim.x + threadIdx.x;
  if (t >= T) return;
  const float* xr = x + (size_t)t * HD;
  float l[NE];
#pragma unroll
  for (int e = 0; e < NE; ++e) l[e] = 0.f;
  for (int k = 0; k < HD; ++k) {
    float xv = xr[k];
#pragma unroll
    for (int e = 0; e < NE; ++e) l[e] += xv * wgs[k * NE + e];
  }
  float mx = l[0];
#pragma unroll
  for (int e = 1; e < NE; ++e) mx = fmaxf(mx, l[e]);
  float p[NE], z = 0.f;
#pragma unroll
  for (int e = 0; e < NE; ++e) { p[e] = __expf(l[e] - mx); z += p[e]; }
  float iz = 1.f / z;
#pragma unroll
  for (int e = 0; e < NE; ++e) p[e] *= iz;
  int i1 = 0; float v1 = p[0];
#pragma unroll
  for (int e = 1; e < NE; ++e) if (p[e] > v1) { i1 = e; v1 = p[e]; }
  int i2 = (i1 == 0) ? 1 : 0; float v2 = p[i2];
#pragma unroll
  for (int e = 0; e < NE; ++e) if (e != i1 && p[e] > v2) { i2 = e; v2 = p[e]; }
  // softmax renorm over the two top probs
  float e2 = __expf(v2 - v1);
  float s  = 1.f + e2;
  ridx[2 * t] = i1;      rw[2 * t] = 1.f / s;
  ridx[2 * t + 1] = i2;  rw[2 * t + 1] = e2 / s;
}

// ------------------------------------------------- deterministic list build
__global__ void moe_build(const int* __restrict__ ridx, const float* __restrict__ rw,
                          int* __restrict__ counts, int* __restrict__ lists,
                          float* __restrict__ wlists) {
  int e = threadIdx.x;
  if (e >= NE) return;
  int c = 0;
  for (int t = 0; t < T; ++t) {
#pragma unroll
    for (int k = 0; k < 2; ++k) {
      if (ridx[2 * t + k] == e) {
        lists[e * T + c]  = t;
        wlists[e * T + c] = rw[2 * t + k];
        ++c;
      }
    }
  }
  counts[e] = c;
}

// ---------------------------------------------------------------- fused FFN
__global__ void __launch_bounds__(NTHR)
moe_ffn(const float* __restrict__ x, const float* __restrict__ W1,
        const float* __restrict__ W2, const int* __restrict__ counts,
        const int* __restrict__ lists, const float* __restrict__ wlists,
        float* __restrict__ out) {
  int e  = blockIdx.x / MBLK;
  int mb = blockIdx.x % MBLK;
  int cnt = counts[e];
  if (mb * MT >= cnt) return;

  extern __shared__ char smem[];
  __bf16* Xs  = (__bf16*)smem;          // [MT][XS]   row-major tokens (A for GEMM1)
  __bf16* W1s = Xs  + MT * XS;          // [FC][W1S]  col-major (B for GEMM1)
  __bf16* W2s = W1s + FC * W1S;         // [HD][W2S]  col-major (B for GEMM2)
  __bf16* Hcs = W2s + HD * W2S;         // [MT][HCS]  row-major (A for GEMM2)
  __shared__ int   toks[MT];
  __shared__ float tws[MT];

  const int tid  = threadIdx.x;
  const int wv   = tid >> 5;   // wave id 0..15
  const int lane = tid & 31;
  const int lm   = lane & 15;
  const int grp  = lane >> 4;  // K-group per ISA 16-bit A/B layout

  if (tid < MT) {
    int i = mb * MT + tid;
    if (i < cnt) { toks[tid] = lists[e * T + i]; tws[tid] = wlists[e * T + i]; }
    else         { toks[tid] = lists[e * T];     tws[tid] = 0.f; }
  }
  __syncthreads();

  // stage X tile (f32 -> bf16), coalesced global reads
  for (int idx = tid; idx < MT * HD; idx += NTHR) {
    int r = idx / HD, k = idx % HD;
    Xs[r * XS + k] = (__bf16)x[(size_t)toks[r] * HD + k];
  }

  const float* W1e = W1 + (size_t)e * HD * FI;
  const float* W2e = W2 + (size_t)e * FI * HD;

  v8f accY[12];
#pragma unroll
  for (int j = 0; j < 12; ++j) accY[j] = (v8f){0.f,0.f,0.f,0.f,0.f,0.f,0.f,0.f};

  for (int fc = 0; fc < FI; fc += FC) {
    __syncthreads();  // previous phase-B done before slabs are overwritten
    // W1 slab [HD x FC] -> col-major bf16 (K pairs contiguous)
    for (int idx = tid; idx < HD * FC; idx += NTHR) {
      int k = idx / FC, n = idx % FC;
      W1s[n * W1S + k] = (__bf16)W1e[(size_t)k * FI + fc + n];
    }
    // W2 slab [FC x HD] -> col-major bf16
    for (int idx = tid; idx < FC * HD; idx += NTHR) {
      int kl = idx / HD, n = idx % HD;
      W2s[n * W2S + kl] = (__bf16)W2e[(size_t)(fc + kl) * HD + n];
    }
    __syncthreads();

    // ---- Phase A: H = silu(X @ W1[:, fc:fc+32]) ; waves 0..7, one 16x16 tile each
    if (wv < 8) {
      int rt = wv & 3, ctH = wv >> 2;   // 4 row-tiles x 2 col-tiles of H
      v8f acc = {};
      for (int kb = 0; kb < HD; kb += 32) {
        v16bf a, b;
#pragma unroll
        for (int v = 0; v < 8; ++v) {
          int kk = kb + ((v & 4) << 2) + grp * 8 + ((v & 3) << 1);
          a[2 * v]     = Xs[(rt * 16 + lm) * XS + kk];
          a[2 * v + 1] = Xs[(rt * 16 + lm) * XS + kk + 1];
          b[2 * v]     = W1s[(ctH * 16 + lm) * W1S + kk];
          b[2 * v + 1] = W1s[(ctH * 16 + lm) * W1S + kk + 1];
        }
        acc = __builtin_amdgcn_wmma_f32_16x16x32_bf16(false, a, false, b,
                                                      (short)0, acc, false, false);
      }
#pragma unroll
      for (int r = 0; r < 8; ++r) {
        float vv = acc[r];
        float sl = vv / (1.f + __expf(-vv));   // silu
        Hcs[(rt * 16 + grp * 8 + r) * HCS + ctH * 16 + lm] = (__bf16)sl;
      }
    }
    __syncthreads();

    // ---- Phase B: Y += H @ W2[fc:fc+32, :] ; all 16 waves, 12 col-tiles each
    {
      int rt = wv & 3, cg = wv >> 2;    // 4 row-tiles x 4 col-groups
      v16bf a;
#pragma unroll
      for (int v = 0; v < 8; ++v) {
        int kk = ((v & 4) << 2) + grp * 8 + ((v & 3) << 1);
        a[2 * v]     = Hcs[(rt * 16 + lm) * HCS + kk];
        a[2 * v + 1] = Hcs[(rt * 16 + lm) * HCS + kk + 1];
      }
#pragma unroll
      for (int j = 0; j < 12; ++j) {
        int n0 = (cg * 12 + j) * 16 + lm;
        v16bf b;
#pragma unroll
        for (int v = 0; v < 8; ++v) {
          int kk = ((v & 4) << 2) + grp * 8 + ((v & 3) << 1);
          b[2 * v]     = W2s[n0 * W2S + kk];
          b[2 * v + 1] = W2s[n0 * W2S + kk + 1];
        }
        accY[j] = __builtin_amdgcn_wmma_f32_16x16x32_bf16(false, a, false, b,
                                                          (short)0, accY[j], false, false);
      }
    }
  }
  __syncthreads();

  // gate-weighted accumulate into output (2 commutative f32 adds per element)
  int rt = wv & 3, cg = wv >> 2;
#pragma unroll
  for (int j = 0; j < 12; ++j) {
    int col = (cg * 12 + j) * 16 + lm;
#pragma unroll
    for (int r = 0; r < 8; ++r) {
      int m = rt * 16 + grp * 8 + r;
      atomicAdd(&out[(size_t)toks[m] * HD + col], accY[j][r] * tws[m]);
    }
  }
}

// ---------------------------------------------------------------- launcher
extern "C" void kernel_launch(void* const* d_in, const int* in_sizes, int n_in,
                              void* d_out, int out_size, void* d_ws, size_t ws_size,
                              hipStream_t stream) {
  (void)in_sizes; (void)n_in; (void)ws_size;
  const float* x  = (const float*)d_in[0];
  const float* Wg = (const float*)d_in[1];
  const float* W1 = (const float*)d_in[2];
  const float* W2 = (const float*)d_in[3];
  float* out = (float*)d_out;

  int*   ridx   = (int*)d_ws;                         // T*2
  float* rw     = (float*)d_ws + 2 * T;               // T*2
  int*   counts = (int*)d_ws + 4 * T;                 // NE (+pad)
  int*   lists  = (int*)d_ws + 4 * T + 16;            // NE*T
  float* wlists = (float*)d_ws + 4 * T + 16 + NE * T; // NE*T

  moe_zero<<<(out_size + 255) / 256, 256, 0, stream>>>(out, out_size);
  moe_route<<<T / 256, 256, 0, stream>>>(x, Wg, ridx, rw);
  moe_build<<<1, NE, 0, stream>>>(ridx, rw, counts, lists, wlists);

  size_t smem = (size_t)(MT * XS + FC * W1S + HD * W2S + MT * HCS) * sizeof(__bf16);
  hipFuncSetAttribute(reinterpret_cast<const void*>(moe_ffn),
                      hipFuncAttributeMaxDynamicSharedMemorySize, (int)smem);
  moe_ffn<<<dim3(NE * MBLK), NTHR, smem, stream>>>(x, W1, W2, counts, lists, wlists, out);
}
